// TransNao_2078764171584
// MI455X (gfx1250) — matrix operand order repaired and checked by reading, not compiled, not run on hardware
//
#include <hip/hip_runtime.h>
#include <hip/hip_bf16.h>

// ---------------------------------------------------------------------------
// Sizes (fixed by the reference)
// ---------------------------------------------------------------------------
#define BB    2
#define LL    2048
#define DD    768
#define HH    8
#define DKK   64
#define DVV   96
#define NPOS  32
#define DFF   1536
#define HDK   (HH * DKK)   // 512
#define HDV   (HH * DVV)   // 768
#define ROWS  (BB * LL)    // 4096

typedef _Float16 v16h __attribute__((ext_vector_type(16)));
typedef _Float16 h8   __attribute__((ext_vector_type(8)));
typedef float    v8f  __attribute__((ext_vector_type(8)));

union H16 { v16h v; h8 h[2]; };

__device__ __forceinline__ v8f wmma16(v16h a, v16h b, v8f c) {
  // D = A(16x32 f16) * B(32x16 f16) + C(16x16 f32)
  return __builtin_amdgcn_wmma_f32_16x16x32_f16(
      /*neg_a=*/false, a, /*neg_b=*/false, b,
      /*c_mod=*/(short)0, c, /*reuse_a=*/false, /*reuse_b=*/false);
}

// A-fragment (16x32 f16): lane m<16 -> K {p..p+7, p+16..p+23}; lanes 16-31
// get the +8 halves via the caller's hi*8 offset baked into p.
__device__ __forceinline__ H16 ldA(const _Float16* p) {
  H16 u;
  u.h[0] = *(const h8*)(p);
  u.h[1] = *(const h8*)(p + 16);
  return u;
}
// B-fragment (32x16 f16): lane%16 = N column; K rows {p..p+7, p+16..p+23} (stride ld).
__device__ __forceinline__ H16 ldB(const _Float16* p, int ld) {
  H16 u;
#pragma unroll
  for (int s = 0; s < 8; ++s) {
    u.v[s] = p[(size_t)s * ld];
    u.v[8 + s] = p[(size_t)(16 + s) * ld];
  }
  return u;
}

// ---------------------------------------------------------------------------
// fp32 -> f16 convert
// ---------------------------------------------------------------------------
__global__ void f32_to_f16_kernel(const float* __restrict__ in,
                                  _Float16* __restrict__ out, int n) {
  int i = blockIdx.x * blockDim.x + threadIdx.x;
  if (i < n) out[i] = (_Float16)in[i];
}

// ---------------------------------------------------------------------------
// RMSNorm (fp32 in) -> f16 out.  One block (256 thr) per row.
// ---------------------------------------------------------------------------
__global__ void rmsnorm_f16_kernel(const float* __restrict__ x,
                                   const float* __restrict__ w,
                                   _Float16* __restrict__ o, int D) {
  __shared__ float red[8];
  int row = blockIdx.x;
  const float* xr = x + (size_t)row * D;
  float ss = 0.f;
  for (int c = threadIdx.x; c < D; c += blockDim.x) { float t = xr[c]; ss += t * t; }
#pragma unroll
  for (int off = 16; off > 0; off >>= 1) ss += __shfl_xor(ss, off, 32);
  int wv = threadIdx.x >> 5;
  if ((threadIdx.x & 31) == 0) red[wv] = ss;
  __syncthreads();
  if (wv == 0) {
    float t = (threadIdx.x < 8) ? red[threadIdx.x] : 0.f;
#pragma unroll
    for (int off = 4; off > 0; off >>= 1) t += __shfl_xor(t, off, 8);
    if (threadIdx.x == 0) red[0] = t;
  }
  __syncthreads();
  float inv = rsqrtf(red[0] / (float)D + 1e-5f);
  for (int c = threadIdx.x; c < D; c += blockDim.x)
    o[(size_t)row * D + c] = (_Float16)(xr[c] * inv * w[c]);
}

// ---------------------------------------------------------------------------
// Generic WMMA GEMM: C[M,N] = A[M,K] @ B[K,N]   (A,B f16 row-major, acc f32)
// Wave tile 32x64.  K stepped by 64 with an explicit, fully named double
// buffer (no runtime-indexed arrays -> no scratch/LDS demotion).
// Requires: M % 32 == 0, N % 64 == 0, K % 64 == 0.
// ---------------------------------------------------------------------------
enum { EPI_F16 = 0, EPI_BIAS_RES_F32 = 1, EPI_BIAS_GELU_F16 = 2 };

__global__ __launch_bounds__(128)
void gemm_wmma_kernel(int M, int N, int K,
                      const _Float16* __restrict__ A, int lda,
                      const _Float16* __restrict__ Bm, int ldb,
                      void* __restrict__ Out, int ldc,
                      const float* __restrict__ bias,
                      const float* __restrict__ resid, int epi) {
  int wid  = blockIdx.x * (blockDim.x >> 5) + (threadIdx.x >> 5);
  int lane = threadIdx.x & 31;
  int mtiles = M >> 5;
  int mt = wid % mtiles, nt = wid / mtiles;
  int row0 = mt << 5, col0 = nt << 6;
  int m = lane & 15, hi = lane >> 4;

  v8f acc[2][4] = {};

  const _Float16* pa0 = A + (size_t)(row0 + m) * lda + hi * 8;
  const _Float16* pa1 = A + (size_t)(row0 + 16 + m) * lda + hi * 8;
  const _Float16* pb  = Bm + (size_t)(hi * 8) * ldb + col0 + m;
  const size_t bstep = (size_t)32 * ldb;

  // prologue: load fragments for kb = 0
  H16 ca0 = ldA(pa0), ca1 = ldA(pa1);
  H16 cb0 = ldB(pb, ldb), cb1 = ldB(pb + 16, ldb), cb2 = ldB(pb + 32, ldb), cb3 = ldB(pb + 48, ldb);

  for (int kb = 0; kb < K; kb += 64) {
    // issue loads for kb+32 ("next" buffer)
    H16 na0 = ldA(pa0 + 32), na1 = ldA(pa1 + 32);
    const _Float16* nb = pb + bstep;
    H16 nb0 = ldB(nb, ldb), nb1 = ldB(nb + 16, ldb), nb2 = ldB(nb + 32, ldb), nb3 = ldB(nb + 48, ldb);

    // 8 WMMAs on "current" while next loads drain
    acc[0][0] = wmma16(ca0.v, cb0.v, acc[0][0]);
    acc[1][0] = wmma16(ca1.v, cb0.v, acc[1][0]);
    acc[0][1] = wmma16(ca0.v, cb1.v, acc[0][1]);
    acc[1][1] = wmma16(ca1.v, cb1.v, acc[1][1]);
    acc[0][2] = wmma16(ca0.v, cb2.v, acc[0][2]);
    acc[1][2] = wmma16(ca1.v, cb2.v, acc[1][2]);
    acc[0][3] = wmma16(ca0.v, cb3.v, acc[0][3]);
    acc[1][3] = wmma16(ca1.v, cb3.v, acc[1][3]);

    pa0 += 64; pa1 += 64; pb += 2 * bstep;

    // issue loads for kb+64 (refill "current"); conditional assign -> phi
    if (kb + 64 < K) {
      ca0 = ldA(pa0); ca1 = ldA(pa1);
      cb0 = ldB(pb, ldb); cb1 = ldB(pb + 16, ldb); cb2 = ldB(pb + 32, ldb); cb3 = ldB(pb + 48, ldb);
    }

    // 8 WMMAs on "next" while refill loads drain
    acc[0][0] = wmma16(na0.v, nb0.v, acc[0][0]);
    acc[1][0] = wmma16(na1.v, nb0.v, acc[1][0]);
    acc[0][1] = wmma16(na0.v, nb1.v, acc[0][1]);
    acc[1][1] = wmma16(na1.v, nb1.v, acc[1][1]);
    acc[0][2] = wmma16(na0.v, nb2.v, acc[0][2]);
    acc[1][2] = wmma16(na1.v, nb2.v, acc[1][2]);
    acc[0][3] = wmma16(na0.v, nb3.v, acc[0][3]);
    acc[1][3] = wmma16(na1.v, nb3.v, acc[1][3]);
  }

  // Epilogue.  C layout: lane(0-15)=col, vgpr r = row r (lanes 16-31: row r+8).
#pragma unroll
  for (int mm = 0; mm < 2; ++mm) {
#pragma unroll
    for (int t = 0; t < 4; ++t) {
      int col = col0 + t * 16 + m;
#pragma unroll
      for (int r = 0; r < 8; ++r) {
        int row = row0 + mm * 16 + r + 8 * hi;
        float v = acc[mm][t][r];
        if (epi == EPI_F16) {
          ((_Float16*)Out)[(size_t)row * ldc + col] = (_Float16)v;
        } else if (epi == EPI_BIAS_RES_F32) {
          ((float*)Out)[(size_t)row * ldc + col] =
              v + bias[col] + resid[(size_t)row * ldc + col];
        } else {
          float u = v + bias[col];
          float gel = 0.5f * u * (1.0f + erff(u * 0.70710678118654752f));
          ((_Float16*)Out)[(size_t)row * ldc + col] = (_Float16)gel;
        }
      }
    }
  }
}

// ---------------------------------------------------------------------------
// k [b,l,h*DK+d] (f16)  ->  kt [b,h,d,l] (f16)   (for contiguous B-fragments)
// ---------------------------------------------------------------------------
__global__ void kt_kernel(const _Float16* __restrict__ kh,
                          _Float16* __restrict__ kt) {
  int idx = blockIdx.x * blockDim.x + threadIdx.x;
  if (idx >= BB * LL * HDK) return;
  int dd = idx & 63;
  int h  = (idx >> 6) & 7;
  int l  = (idx >> 9) & 2047;
  int b  = idx >> 20;
  kt[(((size_t)(b * HH + h)) * DKK + dd) * LL + l] = kh[idx];
}

// ---------------------------------------------------------------------------
// qw = q @ w_pos  then suffix sums over n (unsigned part n<16, signed n>=16).
// SU/SS layout: [((b*L+i)*H + h)*17 + t],  t in [0,16], SU[16]=0.
// ---------------------------------------------------------------------------
__global__ void qw_suffix_kernel(const _Float16* __restrict__ qh,
                                 const float* __restrict__ w_pos,
                                 float* __restrict__ SU, float* __restrict__ SS) {
  int idx = blockIdx.x * blockDim.x + threadIdx.x;   // over B*L*H
  if (idx >= ROWS * HH) return;
  int h = idx & 7;
  int row = idx >> 3;
  const _Float16* qr = qh + (size_t)row * HDK + h * DKK;
  const float* wp = w_pos + (size_t)h * DKK * NPOS;
  float q[DKK];
#pragma unroll 8
  for (int dd = 0; dd < DKK; ++dd) q[dd] = (float)qr[dd];
  float* pu = SU + (size_t)idx * 17;
  float* ps = SS + (size_t)idx * 17;
  pu[16] = 0.f; ps[16] = 0.f;
  float su = 0.f, ss = 0.f;
  for (int t = 15; t >= 0; --t) {
    float a0 = 0.f, a1 = 0.f;
#pragma unroll 8
    for (int dd = 0; dd < DKK; ++dd) {
      a0 += q[dd] * wp[dd * NPOS + t];
      a1 += q[dd] * wp[dd * NPOS + 16 + t];
    }
    su += a0; ss += a1;
    pu[t] = su; ps[t] = ss;
  }
}

// ---------------------------------------------------------------------------
// uk[b,h,j] = dot(u_param[h], k[b,j,h,:])
// ---------------------------------------------------------------------------
__global__ void uk_kernel(const _Float16* __restrict__ kh,
                          const float* __restrict__ u_param,
                          float* __restrict__ UK) {
  int idx = blockIdx.x * blockDim.x + threadIdx.x;   // over B*H*L
  if (idx >= BB * HH * LL) return;
  int j = idx & 2047;
  int h = (idx >> 11) & 7;
  int b = idx >> 14;
  const _Float16* kr = kh + (size_t)(b * LL + j) * HDK + h * DKK;
  float acc = 0.f;
#pragma unroll 8
  for (int dd = 0; dd < DKK; ++dd)
    acc += u_param[h * DKK + dd] * (float)kr[dd];
  UK[idx] = acc;
}

// ---------------------------------------------------------------------------
// vr depends only on (h, d=j-i):  VR[h, d+2047] = sum_n vw[h,n]*basis_n(d)
// ---------------------------------------------------------------------------
__global__ void vr_kernel(const float* __restrict__ v_param,
                          const float* __restrict__ w_pos,
                          float* __restrict__ VR) {
  int idx = blockIdx.x * blockDim.x + threadIdx.x;
  if (idx >= HH * (2 * LL - 1)) return;
  int h  = idx / (2 * LL - 1);
  int di = idx % (2 * LL - 1);
  int d  = di - (LL - 1);
  int ad = d < 0 ? -d : d;
  float sgn = d > 0 ? 1.f : (d < 0 ? -1.f : 0.f);
  float lnr = logf((float)(LL + 1) * 0.5f) / (float)(NPOS / 2);
  float su = 0.f, ss = 0.f;
  for (int nn = 0; nn < 16; ++nn) {
    float wdt = expf((float)(nn + 1) * lnr);
    if ((float)ad <= wdt) {
      float a0 = 0.f, a1 = 0.f;
#pragma unroll 8
      for (int dd = 0; dd < DKK; ++dd) {
        float vp = v_param[h * DKK + dd];
        a0 += vp * w_pos[(size_t)(h * DKK + dd) * NPOS + nn];
        a1 += vp * w_pos[(size_t)(h * DKK + dd) * NPOS + 16 + nn];
      }
      su += a0; ss += a1;
    }
  }
  VR[idx] = su + sgn * ss;
}

// ---------------------------------------------------------------------------
// Flash attention with relative-position bias.
// Block = 256 thr (8 waves); wave w handles 16 query rows of (b,h).
// grid = (L/128, B*H)
// ---------------------------------------------------------------------------
__global__ __launch_bounds__(256)
void attn_kernel(const _Float16* __restrict__ qh,   // [b,l,h*DK]
                 const _Float16* __restrict__ kt,   // [b,h,d,l]
                 const _Float16* __restrict__ vh,   // [b,l,h*DV]
                 const float* __restrict__ SU, const float* __restrict__ SS,
                 const float* __restrict__ UK, const float* __restrict__ VR,
                 _Float16* __restrict__ outh) {     // [b,l,h*DV]
  __shared__ float    s_vr[2 * LL - 1];
  __shared__ float    s_uk[LL];
  __shared__ float    s_su[8][16][17];
  __shared__ float    s_ss[8][16][17];
  __shared__ _Float16 s_p[8][16][32];

  const int bh = blockIdx.y;
  const int b = bh >> 3, h = bh & 7;
  const int w = threadIdx.x >> 5, lane = threadIdx.x & 31;
  const int n = lane & 15, g = lane >> 4;          // C-layout col / row-half
  const int i0 = (blockIdx.x * 8 + w) * 16;

  for (int idx = threadIdx.x; idx < 2 * LL - 1; idx += 256)
    s_vr[idx] = VR[(size_t)h * (2 * LL - 1) + idx];
  for (int idx = threadIdx.x; idx < LL; idx += 256)
    s_uk[idx] = UK[(size_t)bh * LL + idx];
  for (int idx = lane; idx < 16 * 17; idx += 32) {
    int rr = idx / 17, t = idx % 17;
    size_t base = ((size_t)(b * LL + i0 + rr) * HH + h) * 17 + t;
    s_su[w][rr][t] = SU[base];
    s_ss[w][rr][t] = SS[base];
  }
  __syncthreads();

  // resident Q A-fragments (16x64 = two 16x32 fragments)
  const _Float16* qbase = qh + (size_t)(b * LL + i0 + n) * HDK + h * DKK;
  H16 aq0 = ldA(qbase + g * 8);
  H16 aq1 = ldA(qbase + 32 + g * 8);

  v8f oc[6] = {};
  float m_r[8], l_r[8];
#pragma unroll
  for (int r = 0; r < 8; ++r) { m_r[r] = -1e30f; l_r[r] = 0.f; }

  const float inv_log2r = (float)(NPOS / 2) / __log2f((float)(LL + 1) * 0.5f);
  const float scale = 0.125f;    // 1/sqrt(64)
  const _Float16* ktbase = kt + (size_t)bh * DKK * LL;

  for (int j0 = 0; j0 < LL; j0 += 32) {
    // --- issue K-fragment loads first (score WMMAs depend on them) ---
    const _Float16* kp0 = ktbase + (size_t)(g * 8) * LL + j0 + n;
    const _Float16* kp1 = ktbase + (size_t)(32 + g * 8) * LL + j0 + n;
    H16 kf00 = ldB(kp0, LL);        // jt=0, kb2=0
    H16 kf01 = ldB(kp1, LL);        // jt=0, kb2=1
    H16 kf10 = ldB(kp0 + 16, LL);   // jt=1, kb2=0
    H16 kf11 = ldB(kp1 + 16, LL);   // jt=1, kb2=1
    // --- issue V-fragment loads now; they drain during scores + softmax ---
    H16 vf[6];
    const _Float16* vb = vh + (size_t)(b * LL + j0 + g * 8) * HDV + h * DVV + n;
#pragma unroll
    for (int nv = 0; nv < 6; ++nv) {
      vf[nv] = ldB(vb + nv * 16, HDV);
    }

    v8f s[2] = {};
    s[0] = wmma16(aq0.v, kf00.v, s[0]);
    s[0] = wmma16(aq1.v, kf01.v, s[0]);
    s[1] = wmma16(aq0.v, kf10.v, s[1]);
    s[1] = wmma16(aq1.v, kf11.v, s[1]);

    // bias: (qk + qr + uk + vr) * scale
#pragma unroll
    for (int jt = 0; jt < 2; ++jt) {
      int j = j0 + jt * 16 + n;
      float ukj = s_uk[j];
#pragma unroll
      for (int r = 0; r < 8; ++r) {
        int rt = r + 8 * g;
        int i = i0 + rt;
        int d = j - i;
        int ad = d < 0 ? -d : d;
        float sgn = d > 0 ? 1.f : (d < 0 ? -1.f : 0.f);
        int t = 0;
        if (ad > 1) {
          float tf = __log2f((float)ad) * inv_log2r - 1.0f;
          t = (int)ceilf(tf - 1e-4f);
          t = t < 0 ? 0 : (t > 16 ? 16 : t);
        }
        float qr = s_su[w][rt][t] + sgn * s_ss[w][rt][t];
        s[jt][r] = (s[jt][r] + qr + s_vr[d + (LL - 1)] + ukj) * scale;
      }
    }
    // online softmax across this 32-col tile (row = 16-lane group reduction)
#pragma unroll
    for (int r = 0; r < 8; ++r) {
      float mx = fmaxf(s[0][r], s[1][r]);
#pragma unroll
      for (int off = 8; off > 0; off >>= 1) mx = fmaxf(mx, __shfl_xor(mx, off, 16));
      float mnew = fmaxf(m_r[r], mx);
      float corr = __expf(m_r[r] - mnew);
      float p0 = __expf(s[0][r] - mnew);
      float p1 = __expf(s[1][r] - mnew);
      float rs = p0 + p1;
#pragma unroll
      for (int off = 8; off > 0; off >>= 1) rs += __shfl_xor(rs, off, 16);
      m_r[r] = mnew;
      l_r[r] = l_r[r] * corr + rs;
#pragma unroll
      for (int nv = 0; nv < 6; ++nv) oc[nv][r] = oc[nv][r] * corr;
      int rt = r + 8 * g;
      s_p[w][rt][n]      = (_Float16)p0;
      s_p[w][rt][16 + n] = (_Float16)p1;
    }
    asm volatile("s_wait_dscnt 0" ::: "memory");    // P visible across lanes
    // P as A-fragment (16x32); V fragments already in registers
    H16 ap = ldA(&s_p[w][n][g * 8]);
#pragma unroll
    for (int nv = 0; nv < 6; ++nv)
      oc[nv] = wmma16(ap.v, vf[nv].v, oc[nv]);
  }
  // normalize + store f16 [b,l,h*DV]
#pragma unroll
  for (int r = 0; r < 8; ++r) {
    float invl = 1.0f / l_r[r];
    int row = i0 + r + 8 * g;
    size_t obase = (size_t)(b * LL + row) * HDV + h * DVV + n;
#pragma unroll
    for (int nv = 0; nv < 6; ++nv)
      outh[obase + nv * 16] = (_Float16)(oc[nv][r] * invl);
  }
}

// ---------------------------------------------------------------------------
// Host side
// ---------------------------------------------------------------------------
static inline int cdiv(int a, int b) { return (a + b - 1) / b; }

extern "C" void kernel_launch(void* const* d_in, const int* in_sizes, int n_in,
                              void* d_out, int out_size, void* d_ws, size_t ws_size,
                              hipStream_t stream) {
  const float* x       = (const float*)d_in[0];
  const float* norm1_w = (const float*)d_in[1];
  const float* norm2_w = (const float*)d_in[2];
  const float* wq      = (const float*)d_in[3];
  const float* wk      = (const float*)d_in[4];
  const float* wv      = (const float*)d_in[5];
  const float* wo      = (const float*)d_in[6];
  const float* bo      = (const float*)d_in[7];
  const float* u_param = (const float*)d_in[8];
  const float* v_param = (const float*)d_in[9];
  const float* w_pos   = (const float*)d_in[10];
  const float* w1      = (const float*)d_in[11];
  const float* b1      = (const float*)d_in[12];
  const float* w2      = (const float*)d_in[13];
  const float* b2      = (const float*)d_in[14];

  char* base = (char*)d_ws;
  size_t off = 0;
  auto alloc = [&](size_t bytes) -> void* {
    off = (off + 255) & ~(size_t)255;
    void* p = base + off;
    off += bytes;
    return p;
  };

  _Float16* wq_h  = (_Float16*)alloc((size_t)DD * HDK * 2);
  _Float16* wk_h  = (_Float16*)alloc((size_t)DD * HDK * 2);
  _Float16* wv_h  = (_Float16*)alloc((size_t)DD * HDV * 2);
  _Float16* wo_h  = (_Float16*)alloc((size_t)HDV * DD * 2);
  _Float16* w1_h  = (_Float16*)alloc((size_t)DD * DFF * 2);
  _Float16* w2_h  = (_Float16*)alloc((size_t)DFF * DD * 2);
  _Float16* h1_h  = (_Float16*)alloc((size_t)ROWS * DD * 2);
  _Float16* q_h   = (_Float16*)alloc((size_t)ROWS * HDK * 2);
  _Float16* k_h   = (_Float16*)alloc((size_t)ROWS * HDK * 2);
  _Float16* kt_h  = (_Float16*)alloc((size_t)ROWS * HDK * 2);
  _Float16* v_h   = (_Float16*)alloc((size_t)ROWS * HDV * 2);
  float*    SUb   = (float*)alloc((size_t)ROWS * HH * 17 * 4);
  float*    SSb   = (float*)alloc((size_t)ROWS * HH * 17 * 4);
  float*    UKb   = (float*)alloc((size_t)BB * HH * LL * 4);
  float*    VRb   = (float*)alloc((size_t)HH * (2 * LL - 1) * 4);
  _Float16* ao_h  = (_Float16*)alloc((size_t)ROWS * HDV * 2);
  float*    x2    = (float*)alloc((size_t)ROWS * DD * 4);
  _Float16* h2_h  = (_Float16*)alloc((size_t)ROWS * DD * 2);
  _Float16* ff1_h = (_Float16*)alloc((size_t)ROWS * DFF * 2);
  (void)ws_size; (void)in_sizes; (void)n_in; (void)out_size;

  auto conv = [&](const float* src, _Float16* dst, int n) {
    f32_to_f16_kernel<<<cdiv(n, 256), 256, 0, stream>>>(src, dst, n);
  };
  auto gemm = [&](int M, int N, int K, const _Float16* A, int lda,
                  const _Float16* Bm, int ldb, void* Out, int ldc,
                  const float* bias, const float* resid, int epi) {
    int waves = (M / 32) * (N / 64);
    gemm_wmma_kernel<<<waves / 4, 128, 0, stream>>>(M, N, K, A, lda, Bm, ldb,
                                                    Out, ldc, bias, resid, epi);
  };

  // 1) weights -> f16
  conv(wq, wq_h, DD * HDK);
  conv(wk, wk_h, DD * HDK);
  conv(wv, wv_h, DD * HDV);
  conv(wo, wo_h, HDV * DD);
  conv(w1, w1_h, DD * DFF);
  conv(w2, w2_h, DFF * DD);

  // 2) RMSNorm 1
  rmsnorm_f16_kernel<<<ROWS, 256, 0, stream>>>(x, norm1_w, h1_h, DD);

  // 3) QKV projections (WMMA)
  gemm(ROWS, HDK, DD, h1_h, DD, wq_h, HDK, q_h, HDK, nullptr, nullptr, EPI_F16);
  gemm(ROWS, HDK, DD, h1_h, DD, wk_h, HDK, k_h, HDK, nullptr, nullptr, EPI_F16);
  gemm(ROWS, HDV, DD, h1_h, DD, wv_h, HDV, v_h, HDV, nullptr, nullptr, EPI_F16);

  // 4) K transpose pack, bias precomputations
  kt_kernel<<<cdiv(ROWS * HDK, 256), 256, 0, stream>>>(k_h, kt_h);
  qw_suffix_kernel<<<cdiv(ROWS * HH, 128), 128, 0, stream>>>(q_h, w_pos, SUb, SSb);
  uk_kernel<<<cdiv(BB * HH * LL, 256), 256, 0, stream>>>(k_h, u_param, UKb);
  vr_kernel<<<cdiv(HH * (2 * LL - 1), 256), 256, 0, stream>>>(v_param, w_pos, VRb);

  // 5) flash attention with relative bias
  {
    dim3 grid(LL / 128, BB * HH);
    attn_kernel<<<grid, 256, 0, stream>>>(q_h, kt_h, v_h, SUb, SSb, UKb, VRb, ao_h);
  }

  // 6) output projection + bo + residual(x)  -> x2 (fp32)
  gemm(ROWS, DD, HDV, ao_h, HDV, wo_h, DD, x2, DD, bo, x, EPI_BIAS_RES_F32);

  // 7) RMSNorm 2, FFN (GELU exact), final residual -> d_out (fp32)
  rmsnorm_f16_kernel<<<ROWS, 256, 0, stream>>>(x2, norm2_w, h2_h, DD);
  gemm(ROWS, DFF, DD, h2_h, DD, w1_h, DFF, ff1_h, DFF, b1, nullptr, EPI_BIAS_GELU_F16);
  gemm(ROWS, DD, DFF, ff1_h, DFF, w2_h, DD, d_out, DD, b2, x2, EPI_BIAS_RES_F32);
}
